// Attention_v3_21603685499611
// MI455X (gfx1250) — compile-verified, matrix operand order checked by reading, not run on hardware
//
#include <hip/hip_runtime.h>

typedef _Float16 half_t;
typedef __attribute__((ext_vector_type(4)))  _Float16 v4h;
typedef __attribute__((ext_vector_type(8)))  _Float16 v8h;
typedef __attribute__((ext_vector_type(16))) _Float16 v16h;
typedef __attribute__((ext_vector_type(8)))  float    v8f;

#define R_TOTAL 65536   // b*n*r = 4*64*256
#define DIMX    512
#define DHEAD   64
#define SEQ     256
#define EPS_F   1e-5f
#define P_STRIDE 264    // LDS row stride (halves) for P, 16B-aligned, conflict-padded

// Load a 16-bit A/B WMMA fragment (16x32 A, or 32x16 B via its N-major mirror).
// Caller passes p = base + row_or_n*stride + kBase + (lane>>4)*8.
// lane<16 needs K {0..7, 16..23}, lane>=16 needs K {8..15, 24..31}: two b128 loads.
__device__ __forceinline__ v16h frag_ld(const half_t* p) {
  v8h lo = *(const v8h*)(p);
  v8h hi = *(const v8h*)(p + 16);
  return __builtin_shufflevector(lo, hi, 0,1,2,3,4,5,6,7,8,9,10,11,12,13,14,15);
}

__device__ __forceinline__ v8f wmma_f16(v16h a, v16h b, v8f c) {
  return __builtin_amdgcn_wmma_f32_16x16x32_f16(false, a, false, b, (short)0, c, false, false);
}

// CDNA5 async global->LDS copy, 16 bytes per lane, tracked by ASYNCcnt.
__device__ __forceinline__ void async_cp16(const half_t* g, half_t* l) {
  const unsigned ldsa = (unsigned)(uintptr_t)l;  // flat LDS aperture: addr[31:0] = LDS offset
  asm volatile("global_load_async_to_lds_b128 %0, %1, off"
               :: "v"(ldsa), "v"(g) : "memory");
}

// ---------------- Kernel 0a: input LayerNorm, f32 -> f16 ----------------
__global__ __launch_bounds__(256) void ln_in_kernel(const float* __restrict__ x,
                                                    const float* __restrict__ g,
                                                    half_t* __restrict__ xn) {
  const int wave = threadIdx.x >> 5;
  const int lane = threadIdx.x & 31;
  const long row = (long)blockIdx.x * 8 + wave;
  const float* xr = x + row * DIMX;
  float4 v[4], gv[4];
  float s = 0.f, sq = 0.f;
#pragma unroll
  for (int i = 0; i < 4; ++i) {
    v[i]  = ((const float4*)xr)[i * 32 + lane];
    gv[i] = ((const float4*)g)[i * 32 + lane];
    s  += v[i].x + v[i].y + v[i].z + v[i].w;
    sq += v[i].x*v[i].x + v[i].y*v[i].y + v[i].z*v[i].z + v[i].w*v[i].w;
  }
#pragma unroll
  for (int m = 1; m < 32; m <<= 1) { s += __shfl_xor(s, m, 32); sq += __shfl_xor(sq, m, 32); }
  const float mean = s * (1.f / DIMX);
  const float rstd = rsqrtf(sq * (1.f / DIMX) - mean * mean + EPS_F);
#pragma unroll
  for (int i = 0; i < 4; ++i) {
    v4h o;
    o[0] = (half_t)((v[i].x - mean) * rstd * gv[i].x);
    o[1] = (half_t)((v[i].y - mean) * rstd * gv[i].y);
    o[2] = (half_t)((v[i].z - mean) * rstd * gv[i].z);
    o[3] = (half_t)((v[i].w - mean) * rstd * gv[i].w);
    ((v4h*)(xn + row * DIMX))[i * 32 + lane] = o;
  }
}

// ---------------- Kernel 0b: one-time weight transpose+convert (tiny, L2-resident) ----------------
// WqkvT[n][k] = (W_q|W_kv)[k][n] as f16 (640x512); WoutT[n][k] = W_out[k][n] as f16 (512x512)
__global__ __launch_bounds__(256) void prep_w_kernel(const float* __restrict__ Wq,
                                                     const float* __restrict__ Wkv,
                                                     const float* __restrict__ Wout,
                                                     half_t* __restrict__ WqkvT,
                                                     half_t* __restrict__ WoutT) {
  const int idx = blockIdx.x * 256 + threadIdx.x;
  if (idx < 640 * 512) {
    const int n = idx >> 9, k = idx & 511;
    const float w = (n < DIMX) ? Wq[(size_t)k * DIMX + n]
                               : Wkv[(size_t)k * 128 + (n - DIMX)];
    WqkvT[idx] = (half_t)w;
  } else {
    const int i2 = idx - 640 * 512;  // < 512*512
    const int n = i2 >> 9, k = i2 & 511;
    WoutT[i2] = (half_t)Wout[(size_t)k * DIMX + n];
  }
}

// ---------------- Kernel 1: q/k/v projection GEMM  xn @ [W_q | W_kv] ----------------
// grid.x = row blocks of 128, grid.y = 2 halves of N=640 (320 cols each)
#define A_NT 20
__device__ __forceinline__ void stage_proj(const half_t* __restrict__ WqkvT, half_t* buf,
                                           int colBase, int kBase, int tid) {
  // 320 rows x 64 B = 1280 x 16B transfers; 5 async instructions per wave
#pragma unroll
  for (int i = 0; i < 5; ++i) {
    const int e = i * 256 + tid;
    const int nl = e >> 2, part = e & 3;
    async_cp16(WqkvT + (size_t)(colBase + nl) * DIMX + kBase + part * 8,
               &buf[nl * 32 + part * 8]);
  }
}

__global__ __launch_bounds__(256) void proj_kernel(const half_t* __restrict__ xn,
                                                   const half_t* __restrict__ WqkvT,
                                                   half_t* __restrict__ qw,
                                                   half_t* __restrict__ kw,
                                                   half_t* __restrict__ vTw) {
  __shared__ half_t WT[2][320 * 32];  // double-buffered transposed weight chunk (2 x 20 KB)
  const int tid  = threadIdx.x;
  const int wave = tid >> 5, lane = tid & 31;
  const int l = lane & 15, hi = lane >> 4;
  const int rowBase = blockIdx.x * 128 + wave * 16;
  const int colBase = blockIdx.y * 320;
  v8f acc[A_NT];
  v8f z = {};
#pragma unroll
  for (int t = 0; t < A_NT; ++t) acc[t] = z;

  stage_proj(WqkvT, WT[0], colBase, 0, tid);
  for (int kc = 0; kc < 16; ++kc) {
    const int kBase = kc * 32;
    if (kc < 15) {
      stage_proj(WqkvT, WT[(kc + 1) & 1], colBase, kBase + 32, tid);
      asm volatile("s_wait_asynccnt 0x5" ::: "memory");  // older stage complete
    } else {
      asm volatile("s_wait_asynccnt 0x0" ::: "memory");
    }
    __syncthreads();
    const half_t* buf = WT[kc & 1];
    v16h a = frag_ld(xn + (size_t)(rowBase + l) * DIMX + kBase + hi * 8);
#pragma unroll
    for (int t = 0; t < A_NT; ++t) {
      v16h b = frag_ld(&buf[(t * 16 + l) * 32 + hi * 8]);
      acc[t] = wmma_f16(a, b, acc[t]);
    }
    __syncthreads();
  }
  // scatter epilogue: cols [0,512) -> q, [512,576) -> k, [576,640) -> v (transposed)
#pragma unroll
  for (int t = 0; t < A_NT; ++t) {
    const int col = colBase + t * 16 + l;
#pragma unroll
    for (int j = 0; j < 8; ++j) {
      const int row = rowBase + j + hi * 8;
      const half_t hv = (half_t)acc[t][j];
      if (col < DIMX) {
        qw[(size_t)row * DIMX + col] = hv;
      } else if (col < DIMX + DHEAD) {
        kw[(size_t)row * DHEAD + (col - DIMX)] = hv;
      } else {
        const int bn = row >> 8, rl = row & 255, d = col - DIMX - DHEAD;
        vTw[((size_t)bn * DHEAD + d) * SEQ + rl] = hv;  // V^T per (b,n)
      }
    }
  }
}

// ---------------- Kernel 2: fused attention per (b,n,h) ----------------
// block = 256 threads (8 waves); each wave owns 32 rows of the 256x256 score slab.
__global__ __launch_bounds__(256) void attn_kernel(const half_t* __restrict__ qw,
                                                   const half_t* __restrict__ kw,
                                                   const half_t* __restrict__ vTw,
                                                   half_t* __restrict__ mid) {
  extern __shared__ half_t smem[];
  half_t* P   = smem;                       // [256][P_STRIDE] (~132 KB)
  half_t* Ksh = smem + SEQ * P_STRIDE;      // [256][64]  (32 KB)
  half_t* Vsh = Ksh + SEQ * DHEAD;          // [64][256]  (32 KB)
  const int tid  = threadIdx.x;
  const int wave = tid >> 5, lane = tid & 31;
  const int l = lane & 15, hi = lane >> 4;
  const int bn = blockIdx.x >> 3, h = blockIdx.x & 7;
  const half_t* Qg  = qw  + (size_t)bn * SEQ * DIMX + h * DHEAD;   // stride DIMX
  const half_t* Kg  = kw  + (size_t)bn * SEQ * DHEAD;              // [256][64]
  const half_t* VTg = vTw + (size_t)bn * DHEAD * SEQ;              // [64][256]
  const int rBase = wave * 32;

  // async-stage K and V^T into LDS (verbatim copies; 16 async instr per wave)
#pragma unroll
  for (int i = 0; i < 8; ++i) {
    const int e = i * 256 + tid;   // 16B granule index
    async_cp16(Kg  + e * 8, &Ksh[e * 8]);
    async_cp16(VTg + e * 8, &Vsh[e * 8]);
  }

  v8f s[2][16];
  v8f z = {};
#pragma unroll
  for (int rt = 0; rt < 2; ++rt)
#pragma unroll
    for (int ct = 0; ct < 16; ++ct) s[rt][ct] = z;

  // Q A-fragments straight from global (overlaps with the async staging)
  v16h aq[2][2];
#pragma unroll
  for (int kc = 0; kc < 2; ++kc) {
    aq[kc][0] = frag_ld(Qg + (size_t)(rBase + l)      * DIMX + kc * 32 + hi * 8);
    aq[kc][1] = frag_ld(Qg + (size_t)(rBase + 16 + l) * DIMX + kc * 32 + hi * 8);
  }
  asm volatile("s_wait_asynccnt 0x0" ::: "memory");
  __syncthreads();

  // S = Q K^T   (K matrix itself is the N-major mirror of B)
#pragma unroll
  for (int kc = 0; kc < 2; ++kc) {
    const int kBase = kc * 32;
#pragma unroll
    for (int ct = 0; ct < 16; ++ct) {
      v16h b = frag_ld(&Ksh[(ct * 16 + l) * DHEAD + kBase + hi * 8]);
      s[0][ct] = wmma_f16(aq[kc][0], b, s[0][ct]);
      s[1][ct] = wmma_f16(aq[kc][1], b, s[1][ct]);
    }
  }

  // softmax: logits = (s - rowmax) * (SCALE*PB_ALPHA) = (s - rowmax) * 0.125
  float rsum[2][8];
#pragma unroll
  for (int rt = 0; rt < 2; ++rt) {
#pragma unroll
    for (int j = 0; j < 8; ++j) {
      float m = s[rt][0][j];
#pragma unroll
      for (int ct = 1; ct < 16; ++ct) m = fmaxf(m, s[rt][ct][j]);
#pragma unroll
      for (int msk = 1; msk < 16; msk <<= 1) m = fmaxf(m, __shfl_xor(m, msk, 32));
      const int row = rBase + rt * 16 + j + hi * 8;
      float sum = 0.f;
#pragma unroll
      for (int ct = 0; ct < 16; ++ct) {
        const float p = __expf((s[rt][ct][j] - m) * 0.125f);
        sum += p;
        P[(size_t)row * P_STRIDE + ct * 16 + l] = (half_t)p;  // C-layout -> LDS
      }
#pragma unroll
      for (int msk = 1; msk < 16; msk <<= 1) sum += __shfl_xor(sum, msk, 32);
      rsum[rt][j] = sum;
    }
  }
  __syncthreads();

  // O = P @ V  (A-fragments of P re-read from LDS; V^T gives contiguous B-fragments)
  v8f o[2][4];
#pragma unroll
  for (int rt = 0; rt < 2; ++rt)
#pragma unroll
    for (int dt = 0; dt < 4; ++dt) o[rt][dt] = z;
#pragma unroll
  for (int kc = 0; kc < 8; ++kc) {
    const int kBase = kc * 32;
    v16h a0 = frag_ld(&P[(size_t)(rBase + l)      * P_STRIDE + kBase + hi * 8]);
    v16h a1 = frag_ld(&P[(size_t)(rBase + 16 + l) * P_STRIDE + kBase + hi * 8]);
#pragma unroll
    for (int dt = 0; dt < 4; ++dt) {
      v16h b = frag_ld(&Vsh[(dt * 16 + l) * SEQ + kBase + hi * 8]);
      o[0][dt] = wmma_f16(a0, b, o[0][dt]);
      o[1][dt] = wmma_f16(a1, b, o[1][dt]);
    }
  }
  // normalize and store head-concatenated mid
#pragma unroll
  for (int rt = 0; rt < 2; ++rt)
#pragma unroll
    for (int dt = 0; dt < 4; ++dt)
#pragma unroll
      for (int j = 0; j < 8; ++j) {
        const int row = rBase + rt * 16 + j + hi * 8;
        const int col = h * DHEAD + dt * 16 + l;
        mid[((size_t)bn * SEQ + row) * DIMX + col] = (half_t)(o[rt][dt][j] / rsum[rt][j]);
      }
}

// ---------------- Kernel 3: out projection + LayerNorm ----------------
__device__ __forceinline__ void stage_out(const half_t* __restrict__ WoutT, half_t* buf,
                                          int kBase, int tid) {
  // 512 rows x 64 B = 2048 x 16B transfers; 8 async instructions per wave
#pragma unroll
  for (int i = 0; i < 8; ++i) {
    const int e = i * 256 + tid;
    const int nl = e >> 2, part = e & 3;
    async_cp16(WoutT + (size_t)nl * DIMX + kBase + part * 8,
               &buf[nl * 32 + part * 8]);
  }
}

__global__ __launch_bounds__(256) void out_kernel(const half_t* __restrict__ mid,
                                                  const half_t* __restrict__ WoutT,
                                                  const float* __restrict__ gout,
                                                  float* __restrict__ out) {
  __shared__ half_t WT[2][512 * 32];  // double-buffered transposed W_out chunk (2 x 32 KB)
  const int tid  = threadIdx.x;
  const int wave = tid >> 5, lane = tid & 31;
  const int l = lane & 15, hi = lane >> 4;
  const int rowBase = blockIdx.x * 128 + wave * 16;
  v8f acc[32];
  v8f z = {};
#pragma unroll
  for (int t = 0; t < 32; ++t) acc[t] = z;

  stage_out(WoutT, WT[0], 0, tid);
  for (int kc = 0; kc < 16; ++kc) {
    const int kBase = kc * 32;
    if (kc < 15) {
      stage_out(WoutT, WT[(kc + 1) & 1], kBase + 32, tid);
      asm volatile("s_wait_asynccnt 0x8" ::: "memory");  // older stage complete
    } else {
      asm volatile("s_wait_asynccnt 0x0" ::: "memory");
    }
    __syncthreads();
    const half_t* buf = WT[kc & 1];
    v16h a = frag_ld(mid + (size_t)(rowBase + l) * DIMX + kBase + hi * 8);
#pragma unroll
    for (int t = 0; t < 32; ++t) {
      v16h b = frag_ld(&buf[(t * 16 + l) * 32 + hi * 8]);
      acc[t] = wmma_f16(a, b, acc[t]);
    }
    __syncthreads();
  }
  // fused LayerNorm over each 512-wide output row
#pragma unroll
  for (int j = 0; j < 8; ++j) {
    float sm = 0.f, sq = 0.f;
#pragma unroll
    for (int t = 0; t < 32; ++t) { const float vv = acc[t][j]; sm += vv; sq += vv * vv; }
#pragma unroll
    for (int msk = 1; msk < 16; msk <<= 1) {
      sm += __shfl_xor(sm, msk, 32);
      sq += __shfl_xor(sq, msk, 32);
    }
    const float mean = sm * (1.f / DIMX);
    const float rstd = rsqrtf(sq * (1.f / DIMX) - mean * mean + EPS_F);
    const long row = rowBase + j + hi * 8;
#pragma unroll
    for (int t = 0; t < 32; ++t) {
      const int col = t * 16 + l;
      out[row * DIMX + col] = (acc[t][j] - mean) * rstd * gout[col];
    }
  }
}

extern "C" void kernel_launch(void* const* d_in, const int* in_sizes, int n_in,
                              void* d_out, int out_size, void* d_ws, size_t ws_size,
                              hipStream_t stream) {
  const float* x      = (const float*)d_in[0];
  const float* g_norm = (const float*)d_in[1];
  const float* W_q    = (const float*)d_in[2];
  const float* W_kv   = (const float*)d_in[3];
  const float* W_out  = (const float*)d_in[4];
  const float* g_out  = (const float*)d_in[5];
  float* out = (float*)d_out;

  char* ws = (char*)d_ws;
  const size_t xnB = (size_t)R_TOTAL * DIMX * sizeof(half_t);    // 64 MB
  const size_t kB  = (size_t)R_TOTAL * DHEAD * sizeof(half_t);   //  8 MB
  half_t* xn    = (half_t*)ws;                 // reused as `mid` once consumed
  half_t* qw    = (half_t*)(ws + xnB);
  half_t* kw    = (half_t*)(ws + 2 * xnB);
  half_t* vT    = (half_t*)(ws + 2 * xnB + kB);
  half_t* WqkvT = (half_t*)(ws + 2 * xnB + 2 * kB);
  half_t* WoutT = WqkvT + 640 * 512;
  half_t* mid = xn;

  ln_in_kernel<<<R_TOTAL / 8, 256, 0, stream>>>(x, g_norm, xn);
  prep_w_kernel<<<(640 * 512 + 512 * 512) / 256, 256, 0, stream>>>(W_q, W_kv, W_out, WqkvT, WoutT);
  dim3 gA(R_TOTAL / 128, 2);
  proj_kernel<<<gA, 256, 0, stream>>>(xn, WqkvT, qw, kw, vT);
  const size_t attn_lds = (size_t)(SEQ * P_STRIDE + SEQ * DHEAD + DHEAD * SEQ) * sizeof(half_t);
  attn_kernel<<<2048, 256, attn_lds, stream>>>(qw, kw, vT, mid);
  out_kernel<<<R_TOTAL / 128, 256, 0, stream>>>(mid, WoutT, g_out, out);
}
